// HaloBlock_71691594105003
// MI455X (gfx1250) — matrix operand order repaired and checked by reading, not compiled, not run on hardware
//
#include <hip/hip_runtime.h>
#include <hip/hip_bf16.h>

typedef __attribute__((ext_vector_type(16))) _Float16 v16h;
typedef __attribute__((ext_vector_type(8)))  float    v8f;

constexpr int kDIM   = 256;
constexpr int kH     = 128;
constexpr int kW     = 128;
constexpr int kHW    = kH * kW;       // 16384
constexpr int kBS    = 8;
constexpr int kWIN   = 14;            // BS + 2*HALO
constexpr int kDH    = 32;
constexpr int kNKEY  = kWIN * kWIN;   // 196
constexpr int kNKP   = 208;           // 13 * 16 (WMMA N padding)
constexpr int kKP    = 224;           // 7 * 32  (WMMA K padding)
constexpr float kSCALE = 0.17677669529663687f;  // 32^-0.5
constexpr float kEPS   = 1e-5f;

// ---- WMMA fragment loaders per CDNA5 ISA 7.12.2 (wave32) ----
// A 16x32 f16: lane m=lane&15; lane-half h: K = h*16 + {0..15} (contiguous),
// matching lds_b_frag; two f16 per VGPR.
__device__ __forceinline__ v16h lds_a_frag(const _Float16* base, int pitch, int lane) {
  const int h = (lane >> 4) & 1;
  const int m = lane & 15;
  const _Float16* row = base + m * pitch + h * 16;
  v16h a;
#pragma unroll
  for (int i = 0; i < 8; ++i) {
    const int kb = 2 * i;
    a[2 * i]     = row[kb];
    a[2 * i + 1] = row[kb + 1];
  }
  return a;
}
// B 32x16 f16: lane n=lane&15 is the column; lane-half selects K=0..15 / 16..31.
// LDS stores B as [n][k] (contiguous k) so this is 16 contiguous f16.
__device__ __forceinline__ v16h lds_b_frag(const _Float16* base, int pitch, int lane) {
  const _Float16* col = base + (lane & 15) * pitch + (lane >> 4) * 16;
  v16h b;
#pragma unroll
  for (int j = 0; j < 16; ++j) b[j] = col[j];
  return b;
}
// C/D 16x16 f32: VGPR r holds (M = r + 8*(lane>>4), N = lane&15).

// =====================================================================
// Fused 1x1-conv GEMM, double-buffered LDS, tile M=128 x N=128, K-step 32.
// out[b,n,p] = epilogue( sum_k A(in[b,k,p]) * w[n,k] )
// EPI: 0 = identity (f16 out), 1 = relu(bn(acc)) (f16 out),
//      2 = relu(res + gamma*relu(bn(acc))) (f32 out)
// HAS_AT: apply relu(bn_a(x)) to the GEMM input while staging to LDS.
// =====================================================================
template <int EPI, bool HAS_AT, typename TIN, typename TOUT>
__global__ __launch_bounds__(256) void gemm1x1_kernel(
    const TIN* __restrict__ in,
    const float* __restrict__ w0, const float* __restrict__ w1, int n_split,
    const float* __restrict__ at_w, const float* __restrict__ at_b,
    const float* __restrict__ at_m, const float* __restrict__ at_v,
    const float* __restrict__ e_w, const float* __restrict__ e_b,
    const float* __restrict__ e_m, const float* __restrict__ e_v,
    const float* __restrict__ res, const float* __restrict__ gamma,
    TOUT* __restrict__ out, int K, int Nout) {
  __shared__ _Float16 xs[2][128][40];   // A tiles: [m][k]
  __shared__ _Float16 wt[2][128][40];   // B tiles: [n][k]

  const int tid  = threadIdx.x;
  const int lane = tid & 31;
  const int wv   = tid >> 5;                 // 8 waves -> 8 M strips of 16
  const int mb   = blockIdx.x & 127;
  const int bidx = blockIdx.x >> 7;
  const int p0   = mb * 128;
  const int nb   = blockIdx.y * 128;
  const size_t inb = (size_t)bidx * K * kHW;

  v8f acc[8] = {v8f{}, v8f{}, v8f{}, v8f{}, v8f{}, v8f{}, v8f{}, v8f{}};

  const int kk     = tid >> 3;               // 0..31 K row staged by this thread
  const int mstart = (tid & 7) << 4;         // 16 consecutive pixels
  const int wn     = tid >> 1;               // 0..127 weight row
  const int wk0    = (tid & 1) << 4;         // 16 consecutive k
  const int ngl    = nb + wn;
  const float* wrow = (ngl < n_split) ? (w0 + (size_t)ngl * K)
                                      : (w1 + (size_t)(ngl - n_split) * K);

  // ---- prologue: stage K-slice 0 into buffer 0 ----
  {
    float sa = 1.f, ta = 0.f;
    if (HAS_AT) {
      const float s = at_w[kk] * rsqrtf(at_v[kk] + kEPS);
      sa = s;
      ta = at_b[kk] - at_m[kk] * s;
    }
    const TIN* src = in + inb + (size_t)kk * kHW + p0 + mstart;
#pragma unroll
    for (int i = 0; i < 16; ++i) {
      float v = (float)src[i];
      if (HAS_AT) v = fmaxf(v * sa + ta, 0.f);
      xs[0][mstart + i][kk] = (_Float16)v;
    }
#pragma unroll
    for (int i = 0; i < 16; ++i) wt[0][wn][wk0 + i] = (_Float16)wrow[wk0 + i];
  }
  __syncthreads();

  const int NK = K >> 5;
  for (int ks = 0; ks < NK; ++ks) {
    const int cur = ks & 1;
    const int k1  = (ks + 1) << 5;

    // prefetch next K-slice into registers (latency hidden under WMMA)
    TIN   xr[16];
    float wr[16];
    if (k1 < K) {
      const TIN* src = in + inb + (size_t)(k1 + kk) * kHW + p0 + mstart;
#pragma unroll
      for (int i = 0; i < 16; ++i) xr[i] = src[i];
#pragma unroll
      for (int i = 0; i < 16; ++i) wr[i] = wrow[k1 + wk0 + i];
    }

    // 8 WMMAs on the current buffer (one A frag, 8 B frags)
    const v16h a = lds_a_frag(&xs[cur][wv * 16][0], 40, lane);
#pragma unroll
    for (int nt = 0; nt < 8; ++nt) {
      const v16h bfr = lds_b_frag(&wt[cur][nt * 16][0], 40, lane);
      acc[nt] = __builtin_amdgcn_wmma_f32_16x16x32_f16(
          false, a, false, bfr, (short)0, acc[nt], false, false);
    }

    // convert + store prefetched slice into the alternate buffer
    if (k1 < K) {
      float sa = 1.f, ta = 0.f;
      if (HAS_AT) {
        const float s = at_w[k1 + kk] * rsqrtf(at_v[k1 + kk] + kEPS);
        sa = s;
        ta = at_b[k1 + kk] - at_m[k1 + kk] * s;
      }
#pragma unroll
      for (int i = 0; i < 16; ++i) {
        float v = (float)xr[i];
        if (HAS_AT) v = fmaxf(v * sa + ta, 0.f);
        xs[cur ^ 1][mstart + i][kk] = (_Float16)v;
      }
#pragma unroll
      for (int i = 0; i < 16; ++i)
        wt[cur ^ 1][wn][wk0 + i] = (_Float16)wr[i];
    }
    __syncthreads();
  }

  // ---- epilogue ----
  const int h = lane >> 4;
#pragma unroll
  for (int nt = 0; nt < 8; ++nt) {
    const int n = nb + nt * 16 + (lane & 15);
    float s = 0.f, t = 0.f, g = 0.f;
    if (EPI >= 1) {
      const float si = e_w[n] * rsqrtf(e_v[n] + kEPS);
      s = si;
      t = e_b[n] - e_m[n] * si;
    }
    if (EPI == 2) g = gamma[n];
    const size_t ob = ((size_t)bidx * Nout + n) * kHW + p0;
#pragma unroll
    for (int r = 0; r < 8; ++r) {
      const int m = wv * 16 + r + 8 * h;
      float v = acc[nt][r];
      if (EPI == 1) {
        v = fmaxf(v * s + t, 0.f);
      } else if (EPI == 2) {
        const float t2 = fmaxf(v * s + t, 0.f);
        v = fmaxf(res[ob + m] + g * t2, 0.f);
      }
      out[ob + m] = (TOUT)v;
    }
  }
}

// =====================================================================
// Halo attention: one workgroup per (batch*head, 8x8 block).
// q: 64x32, k/v: 196x32 (zero-padded halo window), rel pos via WMMA too.
// =====================================================================
__global__ __launch_bounds__(256) void halo_attn_kernel(
    const _Float16* __restrict__ qkv,     // (b, 768, H, W) f16: q | kv
    const float* __restrict__ relw,       // width_rel  (27, 32)
    const float* __restrict__ relh,       // height_rel (27, 32)
    _Float16* __restrict__ out) {         // (b, 256, H, W) f16
  __shared__ _Float16 q_s[64][32];
  __shared__ _Float16 k_s[kNKP][32];      // [key][d]   (B for QK^T)
  __shared__ _Float16 v_s[32][kKP];       // [d][key]   (B for PV, contiguous k)
  __shared__ _Float16 rw_w[32][32];       // width_rel  [r][d], rows 27..31 zero
  __shared__ _Float16 rh_w[32][32];
  __shared__ float rw_s[64][32];          // q . width_rel[r]
  __shared__ float rh_s[64][32];
  __shared__ _Float16 p_s[64][kKP];       // logits -> probabilities (A for PV)
  __shared__ float redbuf[64][4];
  __shared__ float rmax_s[64];
  __shared__ float rinv_s[64];

  const int tid  = threadIdx.x;
  const int lane = tid & 31;
  const int wv   = tid >> 5;
  const int nb   = blockIdx.x;            // 0..255 spatial block
  const int bh   = blockIdx.y;            // 0..15  batch*head
  const int b    = bh >> 3;
  const int hd   = bh & 7;
  const int bi   = nb >> 4, bj = nb & 15;

  const size_t qbase = (size_t)(b * 768 + hd * 32) * kHW;
  const size_t kbase = (size_t)(b * 768 + 256 + hd * 64) * kHW;
  const size_t vbase = kbase + (size_t)32 * kHW;

  // speculative prefetch of the streaming window (global_prefetch_b8 path)
  __builtin_prefetch(qkv + qbase + (bi * 8) * kW + bj * 8, 0, 0);
  __builtin_prefetch(qkv + kbase + (bi * 8) * kW + bj * 8, 0, 0);
  __builtin_prefetch(qkv + vbase + (bi * 8) * kW + bj * 8, 0, 0);

  // ---- stage q ----
  for (int idx = tid; idx < 32 * 64; idx += 256) {
    const int d = idx >> 6, m = idx & 63;
    const int row = bi * 8 + (m >> 3), col = bj * 8 + (m & 7);
    q_s[m][d] = qkv[qbase + (size_t)d * kHW + row * kW + col];
  }
  // ---- stage k/v halo windows (zero-padded) ----
  for (int idx = tid; idx < 32 * kKP; idx += 256) {
    const int d = idx / kKP;
    const int n = idx - d * kKP;
    _Float16 kval = (_Float16)0.f, vval = (_Float16)0.f;
    if (n < kNKEY) {
      const int i = n / kWIN, j = n - i * kWIN;
      const int row = bi * 8 - 3 + i, col = bj * 8 - 3 + j;
      if (row >= 0 && row < kH && col >= 0 && col < kW) {
        const size_t off = (size_t)d * kHW + row * kW + col;
        kval = qkv[kbase + off];
        vval = qkv[vbase + off];
      }
    }
    if (n < kNKP) k_s[n][d] = kval;
    v_s[d][n] = vval;
  }
  // ---- stage rel tables ----
  for (int idx = tid; idx < 32 * 32; idx += 256) {
    const int r = idx >> 5, d = idx & 31;
    rw_w[r][d] = (r < 27) ? (_Float16)relw[r * 32 + d] : (_Float16)0.f;
    rh_w[r][d] = (r < 27) ? (_Float16)relh[r * 32 + d] : (_Float16)0.f;
  }
  __syncthreads();

  // ---- RW/RH = q @ rel^T (waves 0-3: width, 4-7: height) ----
  {
    const int mt = wv & 3;
    const _Float16* rbase = (wv >> 2) ? &rh_w[0][0] : &rw_w[0][0];
    float* obase = (wv >> 2) ? &rh_s[0][0] : &rw_s[0][0];
    const v16h a = lds_a_frag(&q_s[mt * 16][0], 32, lane);
    const int h = lane >> 4;
#pragma unroll
    for (int nt = 0; nt < 2; ++nt) {
      v8f c = {};
      const v16h bfr = lds_b_frag(rbase + nt * 16 * 32, 32, lane);
      c = __builtin_amdgcn_wmma_f32_16x16x32_f16(false, a, false, bfr, (short)0,
                                                 c, false, false);
      const int n = nt * 16 + (lane & 15);
#pragma unroll
      for (int r = 0; r < 8; ++r) obase[(mt * 16 + r + 8 * h) * 32 + n] = c[r];
    }
  }
  __syncthreads();

  // ---- logits = q @ k^T * SCALE + rel, stored f16 into p_s ----
  {
    const int mt = wv & 3;
    const int h = lane >> 4;
    const v16h a = lds_a_frag(&q_s[mt * 16][0], 32, lane);
    for (int nt = (wv >> 2); nt < 13; nt += 2) {
      const v16h bk = lds_b_frag(&k_s[nt * 16][0], 32, lane);
      v8f c = {};
      c = __builtin_amdgcn_wmma_f32_16x16x32_f16(false, a, false, bk, (short)0,
                                                 c, false, false);
      const int n = nt * 16 + (lane & 15);
      int i = 0, j = 0;
      if (n < kNKEY) { i = n / kWIN; j = n - i * kWIN; }
#pragma unroll
      for (int r = 0; r < 8; ++r) {
        const int m = mt * 16 + r + 8 * h;
        float val;
        if (n < kNKEY) {
          const int rq = m >> 3, cq = m & 7;
          val = c[r] * kSCALE + rw_s[m][j - cq + 13] + rh_s[m][i - rq + 13];
        } else {
          val = -3.0e4f;  // mask (f16-safe)
        }
        p_s[m][n] = (_Float16)val;
      }
    }
  }
  __syncthreads();

  // ---- softmax over 196 keys (4 threads per row) ----
  const int srow = tid >> 2, spart = tid & 3;
  const int ns = spart * 49, ne = ns + 49;
  {
    float mx = -1e30f;
    for (int n = ns; n < ne; ++n) mx = fmaxf(mx, (float)p_s[srow][n]);
    redbuf[srow][spart] = mx;
  }
  __syncthreads();
  if (tid < 64)
    rmax_s[tid] = fmaxf(fmaxf(redbuf[tid][0], redbuf[tid][1]),
                        fmaxf(redbuf[tid][2], redbuf[tid][3]));
  __syncthreads();
  {
    const float rm = rmax_s[srow];
    float sm = 0.f;
    for (int n = ns; n < ne; ++n) {
      const float e = __expf((float)p_s[srow][n] - rm);
      p_s[srow][n] = (_Float16)e;
      sm += e;
    }
    redbuf[srow][spart] = sm;
  }
  // zero padded K columns 196..223 for the PV GEMM
  for (int idx = tid; idx < 64 * (kKP - kNKEY); idx += 256) {
    const int m = idx / (kKP - kNKEY);
    const int n = kNKEY + (idx - m * (kKP - kNKEY));
    p_s[m][n] = (_Float16)0.f;
  }
  __syncthreads();
  if (tid < 64)
    rinv_s[tid] = 1.0f / (redbuf[tid][0] + redbuf[tid][1] + redbuf[tid][2] +
                          redbuf[tid][3]);
  __syncthreads();

  // ---- out = (P @ V) * inv_sum ----
  {
    const int mt = wv & 3, dt = wv >> 2;
    v8f acc = {};
#pragma unroll
    for (int kt = 0; kt < 7; ++kt) {
      const v16h a = lds_a_frag(&p_s[mt * 16][0] + kt * 32, kKP, lane);
      const v16h bv = lds_b_frag(&v_s[dt * 16][0] + kt * 32, kKP, lane);
      acc = __builtin_amdgcn_wmma_f32_16x16x32_f16(false, a, false, bv,
                                                   (short)0, acc, false, false);
    }
    const int h = lane >> 4;
    const int d = dt * 16 + (lane & 15);
    const size_t obase = (size_t)(b * 256 + hd * 32 + d) * kHW;
#pragma unroll
    for (int r = 0; r < 8; ++r) {
      const int m = mt * 16 + r + 8 * h;
      const float o = acc[r] * rinv_s[m];
      const int row = bi * 8 + (m >> 3), col = bj * 8 + (m & 7);
      out[obase + row * kW + col] = (_Float16)o;
    }
  }
}

extern "C" void kernel_launch(void* const* d_in, const int* in_sizes, int n_in,
                              void* d_out, int out_size, void* d_ws,
                              size_t ws_size, hipStream_t stream) {
  const float* x    = (const float*)d_in[0];
  const float* w1   = (const float*)d_in[1];
  const float* q_w  = (const float*)d_in[2];
  const float* kv_w = (const float*)d_in[3];
  const float* w3   = (const float*)d_in[4];
  const float* hrel = (const float*)d_in[5];
  const float* wrel = (const float*)d_in[6];
  const float* bn1w = (const float*)d_in[7];
  const float* bn1b = (const float*)d_in[8];
  const float* bn1m = (const float*)d_in[9];
  const float* bn1v = (const float*)d_in[10];
  const float* bnaw = (const float*)d_in[11];
  const float* bnab = (const float*)d_in[12];
  const float* bnam = (const float*)d_in[13];
  const float* bnav = (const float*)d_in[14];
  const float* bn3w = (const float*)d_in[15];
  const float* bn3b = (const float*)d_in[16];
  const float* bn3m = (const float*)d_in[17];
  const float* bn3v = (const float*)d_in[18];
  const float* gamma = (const float*)d_in[19];
  float* outp = (float*)d_out;

  char* wsb = (char*)d_ws;
  const size_t y1_bytes  = (size_t)2 * 256 * kHW * sizeof(_Float16);  // 16 MB
  const size_t qkv_bytes = (size_t)2 * 768 * kHW * sizeof(_Float16);  // 48 MB
  _Float16* y1  = (_Float16*)wsb;
  _Float16* qkv = (_Float16*)(wsb + y1_bytes);
  _Float16* att = (_Float16*)(wsb + y1_bytes + qkv_bytes);

  dim3 blk(256);
  // Stage 1: y1 = relu(bn1(x @ w1^T))  [M=2*16384, N=256, K=256] -> f16
  gemm1x1_kernel<1, false, float, _Float16><<<dim3(256, 2), blk, 0, stream>>>(
      x, w1, w1, 1 << 30, nullptr, nullptr, nullptr, nullptr, bn1w, bn1b, bn1m,
      bn1v, nullptr, nullptr, y1, 256, 256);
  // Stage 2: qkv = y1 @ [q_w; kv_w]^T  [N=768] -> f16
  gemm1x1_kernel<0, false, _Float16, _Float16>
      <<<dim3(256, 6), blk, 0, stream>>>(y1, q_w, kv_w, 256, nullptr, nullptr,
                                         nullptr, nullptr, nullptr, nullptr,
                                         nullptr, nullptr, nullptr, nullptr,
                                         qkv, 256, 768);
  // Stage 3: halo attention per (batch*head, block)
  halo_attn_kernel<<<dim3(256, 16), blk, 0, stream>>>(qkv, wrel, hrel, att);
  // Stage 4: out = relu(x + gamma*relu(bn3( relu(bna(att)) @ w3^T )))
  gemm1x1_kernel<2, true, _Float16, float><<<dim3(256, 2), blk, 0, stream>>>(
      att, w3, w3, 1 << 30, bnaw, bnab, bnam, bnav, bn3w, bn3b, bn3m, bn3v, x,
      gamma, outp, 256, 256);
}
